// IntegratedModel_57286273794408
// MI455X (gfx1250) — matrix operand order repaired and checked by reading, not compiled, not run on hardware
//
#include <hip/hip_runtime.h>
#include <hip/hip_bf16.h>
#include <math.h>

// ---------------------------------------------------------------------------
// Problem constants (from reference setup_inputs):
//   T=2048, d=512, d_c=256, d_cp=128, d_h=64, r=32, H=8, K_MAX=10, buckets=100
// ---------------------------------------------------------------------------

typedef float        v2f   __attribute__((ext_vector_type(2)));
typedef float        v8f   __attribute__((ext_vector_type(8)));
typedef _Float16     v16h  __attribute__((ext_vector_type(16)));
typedef unsigned int u32x4 __attribute__((ext_vector_type(4)));
typedef int          i32x4 __attribute__((ext_vector_type(4)));
typedef int          i32x8 __attribute__((ext_vector_type(8)));

// Confirmed in round 1: V_WMMA_F32_16X16X4_F32 lowers on gfx1250.
#if defined(__HIP_DEVICE_COMPILE__) && __has_builtin(__builtin_amdgcn_wmma_f32_16x16x4_f32)
#define HAVE_WMMA_F32_DEV 1
#else
#define HAVE_WMMA_F32_DEV 0
#endif

// Tensor Data Mover (async tensor DMA into LDS) + its wait counter.
#if defined(__HIP_DEVICE_COMPILE__) && __has_builtin(__builtin_amdgcn_tensor_load_to_lds) && \
    __has_builtin(__builtin_amdgcn_s_wait_tensorcnt)
#define HAVE_TDM_DEV 1
#else
#define HAVE_TDM_DEV 0
#endif

// ---------------------------------------------------------------------------
// Batched WMMA GEMM, C[b] = op(A[b]) * op(B[b]) (+ bias) (+ cos epilogue).
//   PREC: 0 = fp32 operands (V_WMMA_F32_16X16X4_F32), hash-critical path
//         1 = f16 operands / f32 accum (V_WMMA_F32_16X16X32_F16), value path
//   TA/TB: operand transposes (compile-time => branch-free strided loops)
//   EPI : 1 = escale*cos(acc+bias) (RFF feature map)
//   TDM : 1 = stage the 16xK A row-block into LDS with TENSOR_LOAD_TO_LDS
//         (requires K<=32, TA=0, PREC=0; used for the S_tilde GEMM where the
//          A tile is reused by all N-tiles of a row-block)
// One wave computes a 16x32 C tile (two accumulators share the A fragment).
// grid = (N/32, M/16, batch).  Requires M%16==0, N%32==0, K%32==0 (all hold).
// Fragment layouts per CDNA5 ISA 7.12.2 (wave32), half = lane>>4:
//   A 16xK : lane holds M=lane&15; K-halves split across lane groups
//   B Kx16 : lane holds N=lane&15; same half-group K split
//   C/D    : acc[i] -> row m0 + i + 8*half, col = n0 + (lane&15)
// ---------------------------------------------------------------------------
template <int PREC, int TA, int TB, int EPI, int TDM = 0>
__global__ __launch_bounds__(32)
void gemm_wmma(const float* __restrict__ A, long long sA, int lda,
               const float* __restrict__ B, long long sB, int ldb,
               float* __restrict__ C, long long sC, int ldc,
               int K,
               const float* __restrict__ bias, long long sBias,
               float escale)
{
    const int nb = blockIdx.x, mb = blockIdx.y, bz = blockIdx.z;
    A += (long long)bz * sA;
    B += (long long)bz * sB;
    C += (long long)bz * sC;
    if (bias) bias += (long long)bz * sBias;

    const int lane = threadIdx.x;
    const int half = lane >> 4;     // half-lane-group (K split)
    const int lm   = lane & 15;
    const int row  = mb * 16 + lm;  // A row this lane feeds
    const int col0 = nb * 32 + lm;  // first C column
    const int col1 = col0 + 16;     // second C column

    v8f acc0 = {0.f, 0.f, 0.f, 0.f, 0.f, 0.f, 0.f, 0.f};
    v8f acc1 = {0.f, 0.f, 0.f, 0.f, 0.f, 0.f, 0.f, 0.f};

    if constexpr (PREC == 0) {
#if HAVE_WMMA_F32_DEV
        const float* bP0 = TB ? (B + (long long)col0 * ldb + 2 * half)
                              : (B + (long long)(2 * half) * ldb + col0);
        const float* bP1 = TB ? (B + (long long)col1 * ldb + 2 * half)
                              : (B + (long long)(2 * half) * ldb + col1);
        const long long bStep = TB ? 4 : (long long)4 * ldb;

#if HAVE_TDM_DEV
        if constexpr (TDM && !TA) {
            // ---- Tensor Data Mover: DMA the 16xK fp32 row-block into LDS.
            // D# per CDNA5 ISA 8.3-8.6: 2D tile, data_size=4B, dims == tile.
            __shared__ float aT[16 * 32];                 // K <= 32
            const unsigned ldsOff = (unsigned)(unsigned long long)(size_t)aT;
            const unsigned long long ga =
                (unsigned long long)(size_t)(A + (long long)mb * 16 * lda);
            u32x4 g0;
            g0[0] = 1u;                                   // count=1 (valid)
            g0[1] = ldsOff;                               // lds_addr
            g0[2] = (unsigned)(ga & 0xFFFFFFFFu);         // global_addr[31:0]
            g0[3] = (unsigned)((ga >> 32) & 0x01FFFFFFu)  // global_addr[56:32]
                    | (2u << 30);                         // type = 2 ("image")
            i32x8 g1;
            g1[0] = (int)(2u << 16);                      // data_size = 4B
            g1[1] = (int)(((unsigned)K & 0xFFFFu) << 16); // tensor_dim0 lo16
            g1[2] = (int)((((unsigned)K >> 16) & 0xFFFFu) // tensor_dim0 hi16
                    | (16u << 16));                       // tensor_dim1 = 16
            g1[3] = (int)(((unsigned)K & 0xFFFFu) << 16); // tile_dim0 = K
            g1[4] = 16;                                   // tile_dim1 = 16
            g1[5] = lda;                                  // dim0_stride lo32
            g1[6] = 0;
            g1[7] = 0;
            i32x4 gz4 = {0, 0, 0, 0};
#if __clang_major__ >= 23
            i32x8 gz8 = {0, 0, 0, 0, 0, 0, 0, 0};
            __builtin_amdgcn_tensor_load_to_lds(g0, g1, gz4, gz4, gz8, 0);
#else
            __builtin_amdgcn_tensor_load_to_lds(g0, g1, gz4, gz4, 0);
#endif
            __builtin_amdgcn_s_wait_tensorcnt(0);         // single-wave WG

            const float* aL = aT + (long long)lm * K + 2 * half;
            for (int k0 = 0; k0 < K; k0 += 4) {
                v2f a, b0, b1;
                a.x  = aL[0];
                a.y  = aL[1];
                b0.x = bP0[0]; b0.y = TB ? bP0[1] : bP0[ldb];
                b1.x = bP1[0]; b1.y = TB ? bP1[1] : bP1[ldb];
                acc0 = __builtin_amdgcn_wmma_f32_16x16x4_f32(
                    false, a, false, b0, (short)0, acc0, false, false);
                acc1 = __builtin_amdgcn_wmma_f32_16x16x4_f32(
                    false, a, false, b1, (short)0, acc1, false, false);
                aL += 4; bP0 += bStep; bP1 += bStep;
            }
        } else
#endif
        {
            // fp32: K step 4; this lane's K base is 2*half.
            const float* aP = TA ? (A + (long long)(2 * half) * lda + row)
                                 : (A + (long long)row * lda + 2 * half);
            const long long aStep = TA ? (long long)4 * lda : 4;
            for (int k0 = 0; k0 < K; k0 += 4) {
                __builtin_prefetch(aP + 4 * aStep, 0, 3);   // global_prefetch
                v2f a, b0, b1;
                a.x  = aP[0];
                a.y  = TA ? aP[lda] : aP[1];
                b0.x = bP0[0]; b0.y = TB ? bP0[1] : bP0[ldb];
                b1.x = bP1[0]; b1.y = TB ? bP1[1] : bP1[ldb];
                acc0 = __builtin_amdgcn_wmma_f32_16x16x4_f32(
                    false, a, false, b0, (short)0, acc0, false, false);
                acc1 = __builtin_amdgcn_wmma_f32_16x16x4_f32(
                    false, a, false, b1, (short)0, acc1, false, false);
                aP += aStep; bP0 += bStep; bP1 += bStep;
            }
        }
#endif
    } else {
        // f16: K step 32; A elem j -> K = k0 + 16*half + j
        //                 B elem j -> K = k0 + 16*half + j
        const float* aP  = TA ? (A + (long long)(16 * half) * lda + row)
                              : (A + (long long)row * lda + 16 * half);
        const float* bP0 = TB ? (B + (long long)col0 * ldb + 16 * half)
                              : (B + (long long)(16 * half) * ldb + col0);
        const float* bP1 = TB ? (B + (long long)col1 * ldb + 16 * half)
                              : (B + (long long)(16 * half) * ldb + col1);
        const long long aStep = TA ? (long long)32 * lda : 32;
        const long long bStep = TB ? 32 : (long long)32 * ldb;
        for (int k0 = 0; k0 < K; k0 += 32) {
            __builtin_prefetch(aP + 2 * aStep, 0, 3);       // global_prefetch
            v16h a, b0, b1;
            #pragma unroll
            for (int j = 0; j < 16; ++j) {
                const int dk = ((j >= 8) ? 16 : 0) + (j & 7);
                a[j]  = (_Float16)(TA ? aP[(long long)j * lda] : aP[j]);
                b0[j] = (_Float16)(TB ? bP0[j] : bP0[(long long)j * ldb]);
                b1[j] = (_Float16)(TB ? bP1[j] : bP1[(long long)j * ldb]);
            }
            acc0 = __builtin_amdgcn_wmma_f32_16x16x32_f16(
                false, a, false, b0, (short)0, acc0, false, false);
            acc1 = __builtin_amdgcn_wmma_f32_16x16x32_f16(
                false, a, false, b1, (short)0, acc1, false, false);
            aP += aStep; bP0 += bStep; bP1 += bStep;
        }
    }

    const float bv0 = bias ? bias[col0] : 0.f;
    const float bv1 = bias ? bias[col1] : 0.f;
    #pragma unroll
    for (int i = 0; i < 8; ++i) {
        const int r = mb * 16 + i + 8 * half;
        float v0 = acc0[i] + bv0;
        float v1 = acc1[i] + bv1;
        if constexpr (EPI == 1) { v0 = escale * __cosf(v0); v1 = escale * __cosf(v1); }
        C[(long long)r * ldc + col0] = v0;
        C[(long long)r * ldc + col1] = v1;
    }
}

// ---------------------------------------------------------------------------
// LSH hash: h = mod(int32(dot(sign(P[h,t,:]), r_lsh)), 100)  (trunc-toward-0
// int cast + nonnegative mod, matching astype(int32)+jnp.mod). One thread
// per (h,t); 8*2048 = 16384 rows of 64.
// ---------------------------------------------------------------------------
__global__ __launch_bounds__(256)
void lsh_hash_kernel(const float* __restrict__ P, const float* __restrict__ r_lsh,
                     int* __restrict__ out)
{
    const int i = blockIdx.x * 256 + threadIdx.x;  // 0..16383
    const float* p = P + (long long)i * 64;
    float proj = 0.f;
    #pragma unroll 8
    for (int d = 0; d < 64; ++d) {
        const float x = p[d];
        const float s = (x > 0.f) ? 1.f : ((x < 0.f) ? -1.f : 0.f);
        proj += s * r_lsh[d];
    }
    int q = (int)proj;          // trunc toward zero
    int m = q % 100;
    if (m < 0) m += 100;
    out[i] = m;
}

// ---------------------------------------------------------------------------
// Bucketed top-10 by RFF score, matching jax.lax.top_k tie-breaking
// (descending value, lower index wins ties). Non-bucket entries score -1e30
// exactly as the reference's masked array; `valid` is recovered downstream.
// One wave per (h,t) row: per-lane sorted top-10 over strided j, then a
// 32-way merge in LDS by lane 0.
// ---------------------------------------------------------------------------
__global__ __launch_bounds__(32)
void topk_kernel(const float* __restrict__ S, const int* __restrict__ hq,
                 const int* __restrict__ hk,
                 float* __restrict__ tv, int* __restrict__ ti)
{
    const int row  = blockIdx.x;          // h*2048 + t
    const int h    = row >> 11;
    const int t    = row & 2047;
    const int lane = threadIdx.x;
    const int myh  = hq[row];
    const int* hkh = hk + (h << 11);
    const float* Srow = S + (long long)t * 2048;

    float v[10]; int id[10];
    #pragma unroll
    for (int k = 0; k < 10; ++k) { v[k] = -INFINITY; id[k] = 0x7fffffff; }

    for (int j = lane; j < 2048; j += 32) {
        const float val = (hkh[j] == myh) ? Srow[j] : -1e30f;
        if (val > v[9] || (val == v[9] && j < id[9])) {
            int k = 9;
            while (k > 0 && (val > v[k - 1] || (val == v[k - 1] && j < id[k - 1]))) {
                v[k] = v[k - 1]; id[k] = id[k - 1]; --k;
            }
            v[k] = val; id[k] = j;
        }
    }

    __shared__ float sv[32][10];
    __shared__ int   si[32][10];
    #pragma unroll
    for (int k = 0; k < 10; ++k) { sv[lane][k] = v[k]; si[lane][k] = id[k]; }
    __syncthreads();

    if (lane == 0) {
        int ptr[32];
        for (int l = 0; l < 32; ++l) ptr[l] = 0;
        for (int k = 0; k < 10; ++k) {
            float best = -INFINITY; int bi = 0x7fffffff; int bl = 0;
            for (int l = 0; l < 32; ++l) {
                const float cv = sv[l][ptr[l]];
                const int   ci = si[l][ptr[l]];
                if (cv > best || (cv == best && ci < bi)) { best = cv; bi = ci; bl = l; }
            }
            tv[(long long)row * 10 + k] = best;
            ti[(long long)row * 10 + k] = bi;
            ptr[bl]++;
        }
    }
}

// ---------------------------------------------------------------------------
// 10-key softmax attention. One wave per (h,t); lanes cover d = lane, lane+32.
// valid = rff_val > -5e29 (reference: vals > NEG*0.5). Empty rows -> zeros.
// ---------------------------------------------------------------------------
__global__ __launch_bounds__(32)
void attn_kernel(const float* __restrict__ Qp, const float* __restrict__ Kp,
                 const float* __restrict__ Vp,
                 const float* __restrict__ tv, const int* __restrict__ ti,
                 float* __restrict__ Oh)
{
    const int row  = blockIdx.x;          // h*2048 + t
    const int lane = threadIdx.x;
    const long long hbase = (long long)(row >> 11) * 2048 * 64;
    const float* q = Qp + (long long)row * 64;
    const float q0 = q[lane], q1 = q[lane + 32];

    float logits[10]; int idx[10]; bool valid[10];
    #pragma unroll
    for (int k = 0; k < 10; ++k) {
        const float rv = tv[(long long)row * 10 + k];
        idx[k]   = ti[(long long)row * 10 + k];
        valid[k] = rv > -5e29f;
        const float* kp = Kp + hbase + (long long)idx[k] * 64;
        float part = q0 * kp[lane] + q1 * kp[lane + 32];
        #pragma unroll
        for (int off = 16; off > 0; off >>= 1) part += __shfl_xor(part, off, 32);
        logits[k] = valid[k] ? part * 0.125f : -1e30f;   // 1/sqrt(64)
    }

    float o0 = 0.f, o1 = 0.f;
    float m = -INFINITY; bool any = false;
    #pragma unroll
    for (int k = 0; k < 10; ++k)
        if (valid[k]) { any = true; m = fmaxf(m, logits[k]); }

    if (any) {
        float a[10]; float den = 0.f;
        #pragma unroll
        for (int k = 0; k < 10; ++k) {
            a[k] = valid[k] ? __expf(logits[k] - m) : 0.f;
            den += a[k];
        }
        const float inv = 1.f / den;
        #pragma unroll
        for (int k = 0; k < 10; ++k) {
            if (valid[k]) {
                const float* vp = Vp + hbase + (long long)idx[k] * 64;
                const float w = a[k] * inv;
                o0 += w * vp[lane];
                o1 += w * vp[lane + 32];
            }
        }
    }
    float* o = Oh + (long long)row * 64;
    o[lane] = o0;
    o[lane + 32] = o1;
}

// ---------------------------------------------------------------------------
// Host launcher. Pipeline (all on `stream`):
//   1)  C_Q   = X @ W_DQ_w.T + b                  [2048,128]   f32 WMMA
//   2)  C_KV  = X @ W_DKV_w.T + b                 [2048,256]   f32 WMMA
//   3-8) per-head low-rank Q/K/V projections      [8,2048,64]  f32 WMMA
//   9)  tilde = 0.25*cos(C_KV @ omega + phi_b)    [2048,32]    f32 WMMA+cos
//   10) S = tilde @ tilde.T                       [2048,2048]  f32 WMMA + TDM
//   11-12) LSH hashes hq, hk
//   13) bucketed top-10
//   14) 10-key softmax attention -> Oh
//   15) tmp_cat[t, h*32+r] = (Oh_h @ WO_V_h.T)    [2048,256]   f16 WMMA
//   16) BcatT[o, h*32+r] = Wf_h @ WO_U_h          [512,256]    f16 WMMA
//   17) out = tmp_cat @ BcatT.T + Wf_b            [2048,512]   f16 WMMA
// (16/17 algebraically fuse WO_U with Wf: cuts the 8.6-GFLOP final GEMM 16x;
//  BcatT is produced pre-transposed so 17 loads contiguous K fragments.)
// ---------------------------------------------------------------------------
extern "C" void kernel_launch(void* const* d_in, const int* in_sizes, int n_in,
                              void* d_out, int out_size, void* d_ws, size_t ws_size,
                              hipStream_t stream)
{
    (void)in_sizes; (void)n_in; (void)out_size; (void)ws_size;

    const float* X       = (const float*)d_in[0];
    const float* W_DQ_w  = (const float*)d_in[1];
    const float* W_DQ_b  = (const float*)d_in[2];
    const float* W_DKV_w = (const float*)d_in[3];
    const float* W_DKV_b = (const float*)d_in[4];
    const float* UQ_U    = (const float*)d_in[5];
    const float* UQ_V    = (const float*)d_in[6];
    const float* UK_U    = (const float*)d_in[7];
    const float* UK_V    = (const float*)d_in[8];
    const float* UV_U    = (const float*)d_in[9];
    const float* UV_V    = (const float*)d_in[10];
    const float* WO_U    = (const float*)d_in[11];
    const float* WO_V    = (const float*)d_in[12];
    const float* omega   = (const float*)d_in[13];
    const float* phi_b   = (const float*)d_in[14];
    const float* r_lsh   = (const float*)d_in[15];
    const float* Wf_w    = (const float*)d_in[16];
    const float* Wf_b    = (const float*)d_in[17];
    float* out = (float*)d_out;

    // Workspace carve-up (~43 MB of fp32) -------------------------------
    float* ws = (float*)d_ws;
    size_t off = 0;
    auto alloc = [&](size_t n) { float* p = ws + off; off += n; return p; };
    float* C_Q   = alloc((size_t)2048 * 128);
    float* C_KV  = alloc((size_t)2048 * 256);
    float* R     = alloc((size_t)8 * 2048 * 32);   // reused for RQ/RK/RV
    float* Qp    = alloc((size_t)8 * 2048 * 64);
    float* Kp    = alloc((size_t)8 * 2048 * 64);
    float* Vp    = alloc((size_t)8 * 2048 * 64);
    float* tilde = alloc((size_t)2048 * 32);
    float* S     = alloc((size_t)2048 * 2048);
    float* Oh    = alloc((size_t)8 * 2048 * 64);
    float* tmp   = alloc((size_t)2048 * 256);      // [T, H*32] interleaved
    float* BcatT = alloc((size_t)512 * 256);       // [d, H*32]
    float* tv    = alloc((size_t)8 * 2048 * 10);
    int* hq = (int*)alloc((size_t)8 * 2048);
    int* hk = (int*)alloc((size_t)8 * 2048);
    int* ti = (int*)alloc((size_t)8 * 2048 * 10);

    const dim3 blk(32);
    auto grid = [](int N, int M, int B) { return dim3(N / 32, M / 16, B); };

    // 1) C_Q = X @ W_DQ_w.T + W_DQ_b
    gemm_wmma<0,0,1,0><<<grid(128, 2048, 1), blk, 0, stream>>>(
        X, 0, 512,  W_DQ_w, 0, 512,  C_Q, 0, 128,  512, W_DQ_b, 0, 0.f);
    // 2) C_KV = X @ W_DKV_w.T + W_DKV_b
    gemm_wmma<0,0,1,0><<<grid(256, 2048, 1), blk, 0, stream>>>(
        X, 0, 512,  W_DKV_w, 0, 512,  C_KV, 0, 256,  512, W_DKV_b, 0, 0.f);

    // 3-4) Qp_h = (C_Q @ UQ_V_h.T) @ UQ_U_h.T
    gemm_wmma<0,0,1,0><<<grid(32, 2048, 8), blk, 0, stream>>>(
        C_Q, 0, 128,  UQ_V, 32 * 128, 128,  R, 2048 * 32, 32,  128, nullptr, 0, 0.f);
    gemm_wmma<0,0,1,0><<<grid(64, 2048, 8), blk, 0, stream>>>(
        R, 2048 * 32, 32,  UQ_U, 64 * 32, 32,  Qp, 2048 * 64, 64,  32, nullptr, 0, 0.f);
    // 5-6) Kp_h
    gemm_wmma<0,0,1,0><<<grid(32, 2048, 8), blk, 0, stream>>>(
        C_KV, 0, 256,  UK_V, 32 * 256, 256,  R, 2048 * 32, 32,  256, nullptr, 0, 0.f);
    gemm_wmma<0,0,1,0><<<grid(64, 2048, 8), blk, 0, stream>>>(
        R, 2048 * 32, 32,  UK_U, 64 * 32, 32,  Kp, 2048 * 64, 64,  32, nullptr, 0, 0.f);
    // 7-8) Vp_h
    gemm_wmma<0,0,1,0><<<grid(32, 2048, 8), blk, 0, stream>>>(
        C_KV, 0, 256,  UV_V, 32 * 256, 256,  R, 2048 * 32, 32,  256, nullptr, 0, 0.f);
    gemm_wmma<0,0,1,0><<<grid(64, 2048, 8), blk, 0, stream>>>(
        R, 2048 * 32, 32,  UV_U, 64 * 32, 32,  Vp, 2048 * 64, 64,  32, nullptr, 0, 0.f);

    // 9) tilde = sqrt(2/32) * cos(C_KV @ omega + phi_b)
    gemm_wmma<0,0,0,1><<<grid(32, 2048, 1), blk, 0, stream>>>(
        C_KV, 0, 256,  omega, 0, 32,  tilde, 0, 32,  256, phi_b, 0, 0.25f);
    // 10) S = tilde @ tilde.T  (A row-block staged into LDS by the TDM)
    gemm_wmma<0,0,1,0,1><<<grid(2048, 2048, 1), blk, 0, stream>>>(
        tilde, 0, 32,  tilde, 0, 32,  S, 0, 2048,  32, nullptr, 0, 0.f);

    // 11-12) LSH hashes
    lsh_hash_kernel<<<dim3(64), dim3(256), 0, stream>>>(Qp, r_lsh, hq);
    lsh_hash_kernel<<<dim3(64), dim3(256), 0, stream>>>(Kp, r_lsh, hk);

    // 13) bucketed top-10     14) attention
    topk_kernel<<<dim3(16384), dim3(32), 0, stream>>>(S, hq, hk, tv, ti);
    attn_kernel<<<dim3(16384), dim3(32), 0, stream>>>(Qp, Kp, Vp, tv, ti, Oh);

    // 15) tmp[:, h*32:(h+1)*32] = Oh_h @ WO_V_h.T   (sC=32 interleaves heads)
    gemm_wmma<1,0,1,0><<<grid(32, 2048, 8), blk, 0, stream>>>(
        Oh, 2048 * 64, 64,  WO_V, 32 * 64, 64,  tmp, 32, 256,  64, nullptr, 0, 0.f);
    // 16) BcatT[:, h*32:(h+1)*32] = Wf_h @ WO_U_h   ([512,256], pre-transposed)
    gemm_wmma<1,0,0,0><<<grid(32, 512, 8), blk, 0, stream>>>(
        Wf_w, 512, 4096,  WO_U, 512 * 32, 32,  BcatT, 32, 256,  512, nullptr, 0, 0.f);
    // 17) out = tmp @ BcatT.T + Wf_b
    gemm_wmma<1,0,1,0><<<grid(512, 2048, 1), blk, 0, stream>>>(
        tmp, 0, 256,  BcatT, 0, 256,  out, 0, 512,  256, Wf_b, 0, 0.f);
}
